// ProposalModule_47828755808294
// MI455X (gfx1250) — matrix operand order repaired and checked by reading, not compiled, not run on hardware
//
#include <hip/hip_runtime.h>
#include <cstdint>
#include <cstddef>

#define B_    16
#define K_    2048
#define C_    256
#define P_    512
#define S_    16
#define OUT_  119
#define M0_   (B_*P_*S_)   /* 131072 rows for grouped MLP */
#define M1_   (B_*P_)      /* 8192 rows for FC head */
#define K0P_  288          /* 259 padded up to multiple of 32 */
#define RADIUS_ 0.3f
#define EPS_  1e-5f
#define ATILE (128*40)     /* one A tile in LDS (ushorts), 32 cols + pad 8 */

typedef float  v8f   __attribute__((ext_vector_type(8)));
typedef __bf16 v16bf __attribute__((ext_vector_type(16)));

union FragBF { int4 i4[2]; v16bf v; };

__device__ __forceinline__ float bf2f(unsigned short h) {
    return __uint_as_float(((unsigned)h) << 16);
}
__device__ __forceinline__ unsigned short f2bf(float f) {
    unsigned u = __float_as_uint(f);
    u += 0x7FFFu + ((u >> 16) & 1u);   // round-to-nearest-even
    return (unsigned short)(u >> 16);
}

// ---------------------------------------------------------------------------
// 1) Farthest point sampling: one block per batch, dists[] LDS-resident.
// ---------------------------------------------------------------------------
__global__ __launch_bounds__(256)
void fps_kernel(const float* __restrict__ xyz, float* __restrict__ new_xyz) {
    const int b   = blockIdx.x;
    const int tid = threadIdx.x;
    __shared__ float dists[K_];
    __shared__ float rv[256];
    __shared__ int   ri[256];
    __shared__ float cent[3];
    __shared__ int   s_far;

    const float* xb = xyz + (size_t)b * K_ * 3;
    for (int i = tid; i < K_; i += 256) dists[i] = 1e10f;
    if (tid == 0) s_far = 0;
    __syncthreads();

    for (int p = 0; p < P_; ++p) {
        const int far = s_far;
        if (tid == 0) {
            float x = xb[far*3], y = xb[far*3+1], z = xb[far*3+2];
            cent[0] = x; cent[1] = y; cent[2] = z;
            float* nx = new_xyz + ((size_t)b * P_ + p) * 3;
            nx[0] = x; nx[1] = y; nx[2] = z;
        }
        __syncthreads();
        const float cx = cent[0], cy = cent[1], cz = cent[2];
        float bm = -1.f; int bi = 0;
        for (int i = tid; i < K_; i += 256) {
            float dx = xb[i*3]-cx, dy = xb[i*3+1]-cy, dz = xb[i*3+2]-cz;
            float d  = dx*dx + dy*dy + dz*dz;
            float nd = fminf(dists[i], d);
            dists[i] = nd;
            if (nd > bm) { bm = nd; bi = i; }
        }
        rv[tid] = bm; ri[tid] = bi;
        __syncthreads();
        for (int o = 128; o > 0; o >>= 1) {
            if (tid < o) {
                if (rv[tid+o] > rv[tid] ||
                    (rv[tid+o] == rv[tid] && ri[tid+o] < ri[tid])) {
                    rv[tid] = rv[tid+o]; ri[tid] = ri[tid+o];
                }
            }
            __syncthreads();
        }
        if (tid == 0) s_far = ri[0];
        __syncthreads();
    }
}

// ---------------------------------------------------------------------------
// 2) Ball query (ordered first-16 selection via wave32 ballot) + gather into
//    the bf16 A0 matrix (row = (b,p,s), 288 cols: 3 rel-xyz + 256 feat + pad).
// ---------------------------------------------------------------------------
__global__ __launch_bounds__(256)
void ballgather_kernel(const float* __restrict__ xyz,
                       const float* __restrict__ features,
                       const float* __restrict__ new_xyz,
                       unsigned short* __restrict__ A0) {
    const int bp  = blockIdx.x;          // 0..8191
    const int b   = bp >> 9;             // /P_
    const int tid = threadIdx.x;
    __shared__ int   sidx[S_];
    __shared__ int   sfin[S_];
    __shared__ int   scnt;
    __shared__ float snx[3];

    if (tid < 3) snx[tid] = new_xyz[(size_t)bp*3 + tid];
    if (tid == 0) scnt = 0;
    __syncthreads();

    if (tid < 32) {
        const float cx = snx[0], cy = snx[1], cz = snx[2];
        const float* xb = xyz + (size_t)b * K_ * 3;
        int cnt = 0;
        for (int base = 0; base < K_ && cnt < S_; base += 32) {
            const int k = base + tid;
            float dx = xb[k*3]-cx, dy = xb[k*3+1]-cy, dz = xb[k*3+2]-cz;
            const bool pred = (dx*dx + dy*dy + dz*dz) < (RADIUS_*RADIUS_);
            unsigned m = (unsigned)__ballot(pred);     // wave32: low 32 bits
            const int prefix = __popc(m & ((1u << tid) - 1u));
            if (pred && (cnt + prefix) < S_) sidx[cnt + prefix] = k;
            cnt += __popc(m);
        }
        if (tid == 0) scnt = cnt < S_ ? cnt : S_;
    }
    __syncthreads();
    const int cnt = scnt;
    if (tid < S_) {
        int v;
        if (tid < cnt)      v = sidx[tid];
        else if (cnt > 0)   v = sidx[0];
        else                v = K_ - 1;
        sfin[tid] = v;
    }
    __syncthreads();

    const size_t rowbase = (size_t)bp * S_;
    for (int t = tid; t < S_ * K0P_; t += 256) {
        const int s = t / K0P_;
        const int c = t - s * K0P_;
        const int k = sfin[s];
        float v;
        if (c < 3)        v = (xyz[((size_t)b*K_ + k)*3 + c] - snx[c]) * (1.0f / RADIUS_);
        else if (c < 259) v = features[((size_t)b*C_ + (c-3))*K_ + k];
        else              v = 0.f;
        A0[(rowbase + s) * K0P_ + c] = f2bf(v);
    }
}

// ---------------------------------------------------------------------------
// 3) Weight prep: f32 -> bf16, zero-padded to (dr x dc).
// ---------------------------------------------------------------------------
__global__ __launch_bounds__(256)
void prep_weight_kernel(const float* __restrict__ src, int sr, int sc,
                        unsigned short* __restrict__ dst, int dr, int dc) {
    const int i = blockIdx.x * 256 + threadIdx.x;
    if (i >= dr * dc) return;
    const int r = i / dc, c = i - r * dc;
    const float v = (r < sr && c < sc) ? src[r * sc + c] : 0.f;
    dst[i] = f2bf(v);
}

// ---------------------------------------------------------------------------
// 4) WMMA GEMM: Y[M x 128] = act(norm(A)) @ W^T, bf16 in / f32 accum.
//    256 threads (8 waves) per 128x128 tile; full W in LDS; A tile
//    double-buffered in LDS so global loads for kt+1 overlap WMMAs of kt.
//    Also accumulates per-output-channel sum / sumsq for the next BN.
// ---------------------------------------------------------------------------
__global__ __launch_bounds__(256)
void gemm_bn_kernel(const unsigned short* __restrict__ A, int Kt,
                    const unsigned short* __restrict__ Wg,
                    const float* __restrict__ norm,       // layer stats base or null
                    unsigned short* __restrict__ Y,
                    float* __restrict__ stats_out) {
    extern __shared__ char smem[];
    const int Kt8  = Kt >> 3;
    const int WSTR = Kt + 8;                      // conflict-avoiding pad
    unsigned short* sW = (unsigned short*)smem;   // 128 x WSTR
    unsigned short* sA = sW + 128 * WSTR;         // 2 x (128 x 40) ping-pong
    float* sScale = (float*)(sA + 2 * ATILE);
    float* sShift = sScale + 128;
    float* sSum   = sShift + 128;
    float* sSq    = sSum   + 128;

    const int tid  = threadIdx.x;
    const int lane = tid & 31;
    const int wave = tid >> 5;
    const int l15  = lane & 15;
    const int kbA  = (lane >> 4) * 8;    // C-row offset
    const int kbB  = (lane >> 4) * 16;   // B-frag K group
    const int rowBase = blockIdx.x * 128;

    // per-thread tile-staging coordinates (2 x int4 covers 128x32 tile)
    const int tr0 = tid >> 2;            // rows for j=0 / j=1
    const int tr1 = (tid + 256) >> 2;
    const int tc8 = (tid & 3) * 8;

    if (tid < 128) {
        if (norm) { sScale[tid] = norm[256 + tid]; sShift[tid] = norm[384 + tid]; }
        sSum[tid] = 0.f; sSq[tid] = 0.f;
    }
    // stage full weight matrix into LDS (int4 = 8 bf16 per load)
    for (int i = tid; i < 128 * Kt8; i += 256) {
        const int r  = i / Kt8;
        const int c8 = (i - r * Kt8) * 8;
        *(int4*)(sW + r * WSTR + c8) = *(const int4*)(Wg + (size_t)r * Kt + c8);
    }

    auto gload = [&](int kt, int4* st) {
        st[0] = *(const int4*)(A + (size_t)(rowBase + tr0) * Kt + kt * 32 + tc8);
        st[1] = *(const int4*)(A + (size_t)(rowBase + tr1) * Kt + kt * 32 + tc8);
    };
    auto xform_store = [&](int kt, int buf, int4* st) {
#pragma unroll
        for (int j = 0; j < 2; ++j) {
            const int r    = j ? tr1 : tr0;
            const int gcol = kt * 32 + tc8;
            int4 v = st[j];
            if (norm) {
                unsigned short* u = (unsigned short*)&v;
#pragma unroll
                for (int e = 0; e < 8; ++e) {
                    float x = bf2f(u[e]);
                    x = fmaf(x, sScale[gcol + e], sShift[gcol + e]);
                    x = fmaxf(x, 0.f);
                    u[e] = f2bf(x);
                }
            }
            *(int4*)(sA + buf * ATILE + r * 40 + tc8) = v;
        }
    };

    const v8f zero = {0.f,0.f,0.f,0.f,0.f,0.f,0.f,0.f};
    v8f acc[8];
#pragma unroll
    for (int i = 0; i < 8; ++i) acc[i] = zero;

    const int nkt = Kt >> 5;
    int4 stage[2];
    gload(0, stage);
    xform_store(0, 0, stage);
    __syncthreads();                      // W + tile 0 visible

    for (int kt = 0; kt < nkt; ++kt) {
        const int  cur  = kt & 1;
        const bool more = (kt + 1) < nkt;
        // kick off global loads for the next tile; latency hides under WMMAs
        if (more) gload(kt + 1, stage);
        if (kt + 2 < nkt)                 // gfx1250 global_prefetch_b8
            __builtin_prefetch(A + (size_t)(rowBase + tr0) * Kt + (kt + 2) * 32 + tc8, 0, 1);

        FragBF a;
        {
            const unsigned short* ap = sA + cur * ATILE + (wave * 16 + l15) * 40 + kbB;
            a.i4[0] = *(const int4*)(ap);
            a.i4[1] = *(const int4*)(ap + 8);
        }
#pragma unroll
        for (int nt = 0; nt < 8; ++nt) {
            FragBF bfr;
            const unsigned short* bp = sW + (nt * 16 + l15) * WSTR + kt * 32 + kbB;
            bfr.i4[0] = *(const int4*)(bp);
            bfr.i4[1] = *(const int4*)(bp + 8);
            acc[nt] = __builtin_amdgcn_wmma_f32_16x16x32_bf16(
                false, a.v, false, bfr.v, (short)0, acc[nt], false, false);
        }
        if (more) xform_store(kt + 1, cur ^ 1, stage);
        __syncthreads();
    }

    // store raw pre-BN output as bf16 + per-channel stats
#pragma unroll
    for (int nt = 0; nt < 8; ++nt) {
        const int col = nt * 16 + l15;
        float s = 0.f, q = 0.f;
#pragma unroll
        for (int r = 0; r < 8; ++r) {
            const float v = acc[nt][r];
            const int row = rowBase + wave * 16 + kbA + r;
            Y[(size_t)row * 128 + col] = f2bf(v);
            s += v; q += v * v;
        }
        atomicAdd(&sSum[col], s);
        atomicAdd(&sSq[col],  q);
    }
    __syncthreads();
    if (tid < 128) {
        atomicAdd(&stats_out[tid],       sSum[tid]);
        atomicAdd(&stats_out[128 + tid], sSq[tid]);
    }
}

// ---------------------------------------------------------------------------
// 5) Fold BN stats into per-channel scale/shift: y = x*scale + shift.
// ---------------------------------------------------------------------------
__global__ void finalize_stats_kernel(float* __restrict__ st,
                                      const float* __restrict__ g,
                                      const float* __restrict__ be,
                                      float invN) {
    const int c = threadIdx.x;
    const float m  = st[c] * invN;
    const float v  = st[128 + c] * invN - m * m;
    const float rs = rsqrtf(v + EPS_);
    const float sc = g[c] * rs;
    st[256 + c] = sc;
    st[384 + c] = be[c] - m * sc;
}

// ---------------------------------------------------------------------------
// 6) Max over S neighbors of relu(BN(y)) -> feat (bf16).
// ---------------------------------------------------------------------------
__global__ __launch_bounds__(256)
void maxpool_kernel(const unsigned short* __restrict__ Y2,
                    const float* __restrict__ norm,
                    unsigned short* __restrict__ feat) {
    const int g  = blockIdx.x * 256 + threadIdx.x;   // M1_*128 total
    const int bp = g >> 7;
    const int c  = g & 127;
    const float sc = norm[256 + c], sh = norm[384 + c];
    const unsigned short* p = Y2 + (size_t)bp * S_ * 128 + c;
    float mx = 0.f;                                   // relu => lower bound 0
#pragma unroll
    for (int s = 0; s < S_; ++s)
        mx = fmaxf(mx, fmaf(bf2f(p[s * 128]), sc, sh));
    feat[g] = f2bf(mx);
}

// ---------------------------------------------------------------------------
// 7) Final WMMA GEMM: out = relu(BN(H2)) @ w3^T + b3 (+ new_xyz on cols 2:5).
// ---------------------------------------------------------------------------
__global__ __launch_bounds__(256)
void gemm_final_kernel(const unsigned short* __restrict__ A,
                       const unsigned short* __restrict__ Wg,
                       const float* __restrict__ norm,
                       const float* __restrict__ b3,
                       const float* __restrict__ nxyz,
                       float* __restrict__ out) {
    extern __shared__ char smem[];
    const int Kt = 128, WSTR = 136;
    unsigned short* sW = (unsigned short*)smem;
    unsigned short* sA = sW + 128 * WSTR;             // 2 x ATILE ping-pong
    float* sScale = (float*)(sA + 2 * ATILE);
    float* sShift = sScale + 128;

    const int tid  = threadIdx.x;
    const int lane = tid & 31;
    const int wave = tid >> 5;
    const int l15  = lane & 15;
    const int kbA  = (lane >> 4) * 8;
    const int kbB  = (lane >> 4) * 16;
    const int rowBase = blockIdx.x * 128;

    const int tr0 = tid >> 2;
    const int tr1 = (tid + 256) >> 2;
    const int tc8 = (tid & 3) * 8;

    if (tid < 128) { sScale[tid] = norm[256 + tid]; sShift[tid] = norm[384 + tid]; }
    for (int i = tid; i < 128 * 16; i += 256) {
        const int r  = i >> 4;
        const int c8 = (i & 15) * 8;
        *(int4*)(sW + r * WSTR + c8) = *(const int4*)(Wg + (size_t)r * Kt + c8);
    }

    auto gload = [&](int kt, int4* st) {
        st[0] = *(const int4*)(A + (size_t)(rowBase + tr0) * Kt + kt * 32 + tc8);
        st[1] = *(const int4*)(A + (size_t)(rowBase + tr1) * Kt + kt * 32 + tc8);
    };
    auto xform_store = [&](int kt, int buf, int4* st) {
#pragma unroll
        for (int j = 0; j < 2; ++j) {
            const int r    = j ? tr1 : tr0;
            const int gcol = kt * 32 + tc8;
            int4 v = st[j];
            unsigned short* u = (unsigned short*)&v;
#pragma unroll
            for (int e = 0; e < 8; ++e) {
                float x = bf2f(u[e]);
                x = fmaf(x, sScale[gcol + e], sShift[gcol + e]);
                x = fmaxf(x, 0.f);
                u[e] = f2bf(x);
            }
            *(int4*)(sA + buf * ATILE + r * 40 + tc8) = v;
        }
    };

    const v8f zero = {0.f,0.f,0.f,0.f,0.f,0.f,0.f,0.f};
    v8f acc[8];
#pragma unroll
    for (int i = 0; i < 8; ++i) acc[i] = zero;

    int4 stage[2];
    gload(0, stage);
    xform_store(0, 0, stage);
    __syncthreads();

    for (int kt = 0; kt < 4; ++kt) {
        const int  cur  = kt & 1;
        const bool more = (kt + 1) < 4;
        if (more) gload(kt + 1, stage);
        if (kt + 2 < 4)
            __builtin_prefetch(A + (size_t)(rowBase + tr0) * Kt + (kt + 2) * 32 + tc8, 0, 1);

        FragBF a;
        {
            const unsigned short* ap = sA + cur * ATILE + (wave * 16 + l15) * 40 + kbB;
            a.i4[0] = *(const int4*)(ap);
            a.i4[1] = *(const int4*)(ap + 8);
        }
#pragma unroll
        for (int nt = 0; nt < 8; ++nt) {
            FragBF bfr;
            const unsigned short* bp = sW + (nt * 16 + l15) * WSTR + kt * 32 + kbB;
            bfr.i4[0] = *(const int4*)(bp);
            bfr.i4[1] = *(const int4*)(bp + 8);
            acc[nt] = __builtin_amdgcn_wmma_f32_16x16x32_bf16(
                false, a.v, false, bfr.v, (short)0, acc[nt], false, false);
        }
        if (more) xform_store(kt + 1, cur ^ 1, stage);
        __syncthreads();
    }

#pragma unroll
    for (int nt = 0; nt < 8; ++nt) {
        const int col = nt * 16 + l15;
        if (col < OUT_) {
            const float bias = b3[col];
#pragma unroll
            for (int r = 0; r < 8; ++r) {
                const int row = rowBase + wave * 16 + kbA + r;
                float v = acc[nt][r] + bias;
                if (col >= 2 && col < 5) v += nxyz[(size_t)row * 3 + (col - 2)];
                out[(size_t)row * OUT_ + col] = v;
            }
        }
    }
}

// ---------------------------------------------------------------------------
extern "C" void kernel_launch(void* const* d_in, const int* in_sizes, int n_in,
                              void* d_out, int out_size, void* d_ws, size_t ws_size,
                              hipStream_t stream) {
    const float* xyz      = (const float*)d_in[0];
    const float* features = (const float*)d_in[1];
    const float* w_m0 = (const float*)d_in[2];
    const float* g_m0 = (const float*)d_in[3];
    const float* be_m0= (const float*)d_in[4];
    const float* w_m1 = (const float*)d_in[5];
    const float* g_m1 = (const float*)d_in[6];
    const float* be_m1= (const float*)d_in[7];
    const float* w_m2 = (const float*)d_in[8];
    const float* g_m2 = (const float*)d_in[9];
    const float* be_m2= (const float*)d_in[10];
    const float* w1   = (const float*)d_in[11];
    const float* g1   = (const float*)d_in[13];
    const float* be1  = (const float*)d_in[14];
    const float* w2   = (const float*)d_in[15];
    const float* g2   = (const float*)d_in[17];
    const float* be2  = (const float*)d_in[18];
    const float* w3   = (const float*)d_in[19];
    const float* b3   = (const float*)d_in[20];
    float* out = (float*)d_out;

    char* ws = (char*)d_ws;
    size_t off = 0;
    auto take = [&](size_t bytes) -> void* {
        void* p = ws + off;
        off += (bytes + 255) & ~(size_t)255;
        return p;
    };
    unsigned short* A0   = (unsigned short*)take((size_t)M0_ * K0P_ * 2);
    unsigned short* Y0   = (unsigned short*)take((size_t)M0_ * 128 * 2);
    unsigned short* feat = (unsigned short*)take((size_t)M1_ * 128 * 2);
    unsigned short* H1   = (unsigned short*)take((size_t)M1_ * 128 * 2);
    unsigned short* H2   = (unsigned short*)take((size_t)M1_ * 128 * 2);
    float*          nxyz = (float*)take((size_t)M1_ * 3 * 4);
    unsigned short* W0p  = (unsigned short*)take((size_t)128 * K0P_ * 2);
    unsigned short* Wm1p = (unsigned short*)take((size_t)128 * 128 * 2);
    unsigned short* Wm2p = (unsigned short*)take((size_t)128 * 128 * 2);
    unsigned short* Wf1p = (unsigned short*)take((size_t)128 * 128 * 2);
    unsigned short* Wf2p = (unsigned short*)take((size_t)128 * 128 * 2);
    unsigned short* W3p  = (unsigned short*)take((size_t)128 * 128 * 2);
    float*          stats= (float*)take((size_t)5 * 512 * 4);
    // buffer aliasing: A0 dead after gemm0 -> Y1 lives there; Y0 dead after gemm1 -> Y2.
    unsigned short* Y1 = A0;
    unsigned short* Y2 = Y0;

    hipMemsetAsync(stats, 0, 5 * 512 * sizeof(float), stream);

    prep_weight_kernel<<<(128*K0P_ + 255)/256, 256, 0, stream>>>(w_m0, 128, C_+3, W0p, 128, K0P_);
    prep_weight_kernel<<<64, 256, 0, stream>>>(w_m1, 128, 128, Wm1p, 128, 128);
    prep_weight_kernel<<<64, 256, 0, stream>>>(w_m2, 128, 128, Wm2p, 128, 128);
    prep_weight_kernel<<<64, 256, 0, stream>>>(w1,   128, 128, Wf1p, 128, 128);
    prep_weight_kernel<<<64, 256, 0, stream>>>(w2,   128, 128, Wf2p, 128, 128);
    prep_weight_kernel<<<64, 256, 0, stream>>>(w3, OUT_, 128, W3p, 128, 128);

    fps_kernel<<<B_, 256, 0, stream>>>(xyz, nxyz);
    ballgather_kernel<<<M1_, 256, 0, stream>>>(xyz, features, nxyz, A0);

    auto shm = [](int Kt) -> size_t {
        return (size_t)128 * (Kt + 8) * 2 + (size_t)2 * ATILE * 2 + 4 * 128 * sizeof(float);
    };

    gemm_bn_kernel<<<M0_/128, 256, shm(K0P_), stream>>>(A0, K0P_, W0p, nullptr, Y0, stats + 0*512);
    finalize_stats_kernel<<<1, 128, 0, stream>>>(stats + 0*512, g_m0, be_m0, 1.f / M0_);
    gemm_bn_kernel<<<M0_/128, 256, shm(128), stream>>>(Y0, 128, Wm1p, stats + 0*512, Y1, stats + 1*512);
    finalize_stats_kernel<<<1, 128, 0, stream>>>(stats + 1*512, g_m1, be_m1, 1.f / M0_);
    gemm_bn_kernel<<<M0_/128, 256, shm(128), stream>>>(Y1, 128, Wm2p, stats + 1*512, Y2, stats + 2*512);
    finalize_stats_kernel<<<1, 128, 0, stream>>>(stats + 2*512, g_m2, be_m2, 1.f / M0_);
    maxpool_kernel<<<(M1_*128)/256, 256, 0, stream>>>(Y2, stats + 2*512, feat);
    gemm_bn_kernel<<<M1_/128, 256, shm(128), stream>>>(feat, 128, Wf1p, nullptr, H1, stats + 3*512);
    finalize_stats_kernel<<<1, 128, 0, stream>>>(stats + 3*512, g1, be1, 1.f / M1_);
    gemm_bn_kernel<<<M1_/128, 256, shm(128), stream>>>(H1, 128, Wf2p, stats + 3*512, H2, stats + 4*512);
    finalize_stats_kernel<<<1, 128, 0, stream>>>(stats + 4*512, g2, be2, 1.f / M1_);
    gemm_final_kernel<<<M1_/128, 256, shm(128) - 2*128*sizeof(float), stream>>>(H2, W3p, stats + 4*512, b3, nxyz, out);

    (void)in_sizes; (void)n_in; (void)out_size; (void)ws_size;
}
